// RNNModel_52621939310806
// MI455X (gfx1250) — compile-verified, matrix operand order checked by reading, not run on hardware
//
#include <hip/hip_runtime.h>

// ---------------------------------------------------------------------------
// RevGRU + decoder for MI455X (gfx1250, wave32, WMMA).
//   - RNN scan: persistent 1-WG kernel per layer, state in LDS (144KB),
//     GEMMs via v_wmma_f32_16x16x32_bf16 against fragment-packed bf16 weights.
//   - Decoder (dominant, 140 GFLOP + 545MB f32 out): tiled bf16 WMMA GEMM with
//     double-buffered global_load_async_to_lds_b128 staging of the B operand.
// ---------------------------------------------------------------------------

#define S_LEN 128
#define BATCH 32
#define DIM   512
#define H2D   256
#define CATW  768           // D + H2
#define NTOKN 33278
#define NPAD  33280         // NTOK padded to multiple of 16
#define KT_RNN 24           // 768/32 K-tiles
#define KT_DEC 16           // 512/32 K-tiles

typedef __attribute__((ext_vector_type(16))) __bf16        v16bf;
typedef __attribute__((ext_vector_type(8)))  float         v8f;
typedef __attribute__((ext_vector_type(4)))  unsigned int  u32x4;

union ABfrag { u32x4 q[2]; v16bf v; };
static_assert(sizeof(ABfrag) == 32, "frag size");

__device__ inline unsigned int f32_to_bf16(float f) {
  unsigned int u = __float_as_uint(f);
  return ((u + 0x7FFFu + ((u >> 16) & 1u)) >> 16) & 0xFFFFu;   // RNE
}
__device__ inline unsigned short f32_to_bf16s(float f) {
  return (unsigned short)f32_to_bf16(f);
}
__device__ inline float sigmoidf_(float x) { return 1.0f / (1.0f + __expf(-x)); }

// ---------------------------------------------------------------------------
// Pack a row-major W[Nsrc, K] (f32) into bf16 WMMA-B fragment order:
// per (nt,kk) tile: 32 lanes x 8 dwords contiguous. Lane<16: N=lane, K-pairs
// {0..7,16..23}; lane>=16: N=lane-16, K-pairs {8..15,24..31} (16-bit layout).
// ---------------------------------------------------------------------------
__global__ void pack_b_kernel(const float* __restrict__ W, unsigned int* __restrict__ P,
                              int K, int Npad, int Nsrc) {
  int tid = blockIdx.x * blockDim.x + threadIdx.x;
  int total = (K * Npad) >> 1;              // output dwords
  if (tid >= total) return;
  int ktiles = K >> 5;
  int v     = tid & 7;
  int lane  = (tid >> 3) & 31;
  int chunk = tid >> 8;                      // 256 dwords per tile
  int kk    = chunk % ktiles;
  int nt    = chunk / ktiles;
  int n     = nt * 16 + (lane & 15);
  int half  = lane >> 4;
  int kb    = (v < 4) ? (half * 8 + 2 * v) : (16 + half * 8 + 2 * (v - 4));
  int k     = kk * 32 + kb;
  float f0 = 0.0f, f1 = 0.0f;
  if (n < Nsrc) {
    f0 = W[(size_t)n * K + k];
    f1 = W[(size_t)n * K + k + 1];
  }
  P[tid] = f32_to_bf16(f0) | (f32_to_bf16(f1) << 16);
}

// Embedding gather: x0[s,b,:] = bf16(embed[idx[s,b],:]), dword-packed.
__global__ void gather_kernel(const int* __restrict__ idx, const float* __restrict__ embed,
                              unsigned int* __restrict__ x0, int total_dw) {
  int tid = blockIdx.x * blockDim.x + threadIdx.x;
  if (tid >= total_dw) return;
  int row = tid / (DIM / 2);
  int c2  = tid % (DIM / 2);
  int tok = idx[row];
  const float* e = embed + (size_t)tok * DIM + 2 * c2;
  x0[tid] = f32_to_bf16(e[0]) | (f32_to_bf16(e[1]) << 16);
}

// One 16x16 output tile: A from LDS concat buffer (row-major, stride CATW),
// B from fragment-packed global weights. f32 accumulate.
__device__ inline v8f compute_tile(const unsigned short* catBase, int mt, int nt,
                                   const unsigned int* __restrict__ P, int ktiles) {
  const int lane = threadIdx.x & 31;
  const int half = lane >> 4, l15 = lane & 15;
  v8f acc = {0.f, 0.f, 0.f, 0.f, 0.f, 0.f, 0.f, 0.f};
  const unsigned short* arow = catBase + (size_t)(mt * 16 + l15) * CATW;
#pragma unroll 4
  for (int kk = 0; kk < ktiles; ++kk) {
    ABfrag a, b;
    int kb = kk * 32 + half * 8;
    a.q[0] = *(const u32x4*)(arow + kb);        // K = kb .. kb+7
    a.q[1] = *(const u32x4*)(arow + kb + 16);   // K = kb+16 .. kb+23
    const u32x4* bp = (const u32x4*)(P + ((size_t)(nt * ktiles + kk) * 32 + lane) * 8);
    b.q[0] = bp[0];
    b.q[1] = bp[1];
    acc = __builtin_amdgcn_wmma_f32_16x16x32_bf16(false, a.v, false, b.v,
                                                  (short)0, acc, false, false);
  }
  return acc;
}

// ---------------------------------------------------------------------------
// Persistent reversible-GRU layer: one workgroup (16 waves), S-step scan.
// LDS: cat[32][768] bf16 (48K), hbuf[32][512] f32 (64K), zbuf[32][256] f32 (32K).
// ---------------------------------------------------------------------------
__global__ __launch_bounds__(512)
void rnn_layer_kernel(const unsigned int* __restrict__ xin_dw,
                      unsigned int* __restrict__ xout_dw,
                      const float* __restrict__ h0,
                      const unsigned int* __restrict__ Pzr1, const float* __restrict__ bzr1,
                      const unsigned int* __restrict__ Pg1,  const float* __restrict__ bg1,
                      const unsigned int* __restrict__ Pzr2, const float* __restrict__ bzr2,
                      const unsigned int* __restrict__ Pg2,  const float* __restrict__ bg2,
                      float* __restrict__ hT_out) {
  __shared__ unsigned short cat[BATCH][CATW];
  __shared__ float hbuf[BATCH][DIM];
  __shared__ float zbuf[BATCH][H2D];
  const int tid  = threadIdx.x;
  const int lane = tid & 31, wave = tid >> 5;
  const int half = lane >> 4, l15 = lane & 15;

  for (int i = tid; i < BATCH * DIM; i += 512) (&hbuf[0][0])[i] = h0[i];
  __syncthreads();

  for (int s = 0; s < S_LEN; ++s) {
    // Phase A: cat = [x_s, bf16(h2)]; also flush x_out[s-1] = bf16(h).
    for (int i = tid; i < BATCH * (DIM / 2); i += 512) {
      int m = i >> 8, c = i & 255;
      ((unsigned int*)&cat[m][0])[c] = xin_dw[(size_t)(s * BATCH + m) * (DIM / 2) + c];
    }
    for (int i = tid; i < BATCH * H2D; i += 512) {
      int m = i >> 8, j = i & 255;
      cat[m][DIM + j] = f32_to_bf16s(hbuf[m][H2D + j]);
    }
    if (s > 0) {
      for (int i = tid; i < BATCH * (DIM / 2); i += 512) {
        int m = i >> 8, c = i & 255;
        unsigned int lo = f32_to_bf16(hbuf[m][2 * c]);
        unsigned int hi = f32_to_bf16(hbuf[m][2 * c + 1]);
        xout_dw[(size_t)((s - 1) * BATCH + m) * (DIM / 2) + c] = lo | (hi << 16);
      }
    }
    __syncthreads();

    // GEMM1: zr1 = sigmoid([x,h2] @ Wzr1^T + b)   (64 tiles, 4 per wave)
    v8f acc4[4];
#pragma unroll
    for (int i = 0; i < 4; ++i) {
      int t = wave + 16 * i;
      acc4[i] = compute_tile(&cat[0][0], t & 1, t >> 1, Pzr1, KT_RNN);
    }
    __syncthreads();
#pragma unroll
    for (int i = 0; i < 4; ++i) {
      int t = wave + 16 * i;
      int n = (t >> 1) * 16 + l15;
      int mbase = (t & 1) * 16 + half * 8;
#pragma unroll
      for (int r = 0; r < 8; ++r) {
        int m = mbase + r;
        float v = sigmoidf_(acc4[i][r] + bzr1[n]);
        if (n < H2D) zbuf[m][n] = 0.875f * v + 0.125f;                    // z1
        else { int j = n - H2D; cat[m][DIM + j] = f32_to_bf16s(v * hbuf[m][H2D + j]); } // r1*h2
      }
    }
    __syncthreads();

    // GEMM2: g1 = tanh([x, r1*h2] @ Wg1^T + b); h1n = z1*h1 + (1-z1)*g1
    v8f acc2[2];
#pragma unroll
    for (int i = 0; i < 2; ++i) {
      int t = wave + 16 * i;
      acc2[i] = compute_tile(&cat[0][0], t & 1, t >> 1, Pg1, KT_RNN);
    }
    __syncthreads();
#pragma unroll
    for (int i = 0; i < 2; ++i) {
      int t = wave + 16 * i;
      int n = (t >> 1) * 16 + l15;
      int mbase = (t & 1) * 16 + half * 8;
#pragma unroll
      for (int r = 0; r < 8; ++r) {
        int m = mbase + r;
        float g = tanhf(acc2[i][r] + bg1[n]);
        float z = zbuf[m][n];
        float h1n = z * hbuf[m][n] + (1.0f - z) * g;
        hbuf[m][n] = h1n;
        cat[m][DIM + n] = f32_to_bf16s(h1n);          // cat = [x, h1n]
      }
    }
    __syncthreads();

    // GEMM3: zr2 = sigmoid([x,h1n] @ Wzr2^T + b)
#pragma unroll
    for (int i = 0; i < 4; ++i) {
      int t = wave + 16 * i;
      acc4[i] = compute_tile(&cat[0][0], t & 1, t >> 1, Pzr2, KT_RNN);
    }
    __syncthreads();
#pragma unroll
    for (int i = 0; i < 4; ++i) {
      int t = wave + 16 * i;
      int n = (t >> 1) * 16 + l15;
      int mbase = (t & 1) * 16 + half * 8;
#pragma unroll
      for (int r = 0; r < 8; ++r) {
        int m = mbase + r;
        float v = sigmoidf_(acc4[i][r] + bzr2[n]);
        if (n < H2D) zbuf[m][n] = 0.875f * v + 0.125f;                    // z2
        else { int j = n - H2D; cat[m][DIM + j] = f32_to_bf16s(v * hbuf[m][j]); }       // r2*h1n
      }
    }
    __syncthreads();

    // GEMM4: g2 = tanh([x, r2*h1n] @ Wg2^T + b); h2n = z2*h2 + (1-z2)*g2
#pragma unroll
    for (int i = 0; i < 2; ++i) {
      int t = wave + 16 * i;
      acc2[i] = compute_tile(&cat[0][0], t & 1, t >> 1, Pg2, KT_RNN);
    }
    __syncthreads();
#pragma unroll
    for (int i = 0; i < 2; ++i) {
      int t = wave + 16 * i;
      int n = (t >> 1) * 16 + l15;
      int mbase = (t & 1) * 16 + half * 8;
#pragma unroll
      for (int r = 0; r < 8; ++r) {
        int m = mbase + r;
        float g = tanhf(acc2[i][r] + bg2[n]);
        float z = zbuf[m][n];
        hbuf[m][H2D + n] = z * hbuf[m][H2D + n] + (1.0f - z) * g;
      }
    }
    __syncthreads();
  }

  // Flush x_out[S-1] and final hidden state.
  for (int i = tid; i < BATCH * (DIM / 2); i += 512) {
    int m = i >> 8, c = i & 255;
    unsigned int lo = f32_to_bf16(hbuf[m][2 * c]);
    unsigned int hi = f32_to_bf16(hbuf[m][2 * c + 1]);
    xout_dw[(size_t)((S_LEN - 1) * BATCH + m) * (DIM / 2) + c] = lo | (hi << 16);
  }
  for (int i = tid; i < BATCH * DIM; i += 512) hT_out[i] = (&hbuf[0][0])[i];
}

// ---------------------------------------------------------------------------
// Decoder: out[4096, NTOK] = X[4096,512](bf16) @ embed^T + bias.
// Grid (32, 520); 8 waves/WG; wave computes 16 rows x 64 cols (4 N-tiles).
// B operand (4KB per K-step, shared by all 8 waves) is staged into LDS with
// double-buffered global_load_async_to_lds_b128 (ASYNCcnt), so the fill of
// stage k+1 overlaps the 4 WMMAs of stage k; drained with s_wait_asynccnt.
// ---------------------------------------------------------------------------
__global__ __launch_bounds__(256)
void decoder_kernel(const unsigned short* __restrict__ X,   // [4096,512] bf16 row-major
                    const unsigned int* __restrict__ PE,    // packed embed^T, ktiles=16
                    const float* __restrict__ bias,
                    float* __restrict__ out) {
  __shared__ __align__(16) unsigned char bstage[2][4096];
  const int tid  = threadIdx.x;
  const int wave = tid >> 5, lane = tid & 31;
  const int half = lane >> 4, l15 = lane & 15;
  const int mrow = blockIdx.x * 128 + wave * 16 + l15;       // A row for this lane
  const int nt0  = blockIdx.y * 4;                           // first N-tile

  // Staging map: thread tid copies 16B; chunk j = tid>>6 (one 1KB tile chunk
  // per 64 threads), dword-within-chunk = (tid&63)*4.
  const unsigned lds_base = (unsigned)(unsigned long long)(&bstage[0][0]);
  const int jj = tid >> 6;
  const unsigned int* src0 =
      PE + ((size_t)(nt0 + jj) * KT_DEC) * 256 + (tid & 63) * 4;
  const unsigned my_lds = lds_base + (unsigned)tid * 16u;

#define ISSUE_STAGE(kk, buf)                                                   \
  do {                                                                         \
    unsigned long long ga_ =                                                   \
        (unsigned long long)(const void*)(src0 + (size_t)(kk) * 256);          \
    unsigned lo_ = my_lds + (unsigned)(buf) * 4096u;                           \
    asm volatile("global_load_async_to_lds_b128 %0, %1, off"                   \
                 :: "v"(lo_), "v"(ga_) : "memory");                            \
  } while (0)

  v8f acc[4];
#pragma unroll
  for (int j = 0; j < 4; ++j) acc[j] = (v8f){0.f,0.f,0.f,0.f,0.f,0.f,0.f,0.f};

  ISSUE_STAGE(0, 0);
  asm volatile("s_wait_asynccnt 0x0" ::: "memory");
  __syncthreads();

  for (int kk = 0; kk < KT_DEC; ++kk) {
    const int buf = kk & 1;
    if (kk + 1 < KT_DEC) ISSUE_STAGE(kk + 1, buf ^ 1);   // overlaps compute

    ABfrag a;
    int kb = kk * 32 + half * 8;
    const u32x4* ap = (const u32x4*)(X + (size_t)mrow * DIM + kb);
    a.q[0] = ap[0];        // K = kb .. kb+7
    a.q[1] = ap[2];        // K = kb+16 .. kb+23  (+16 shorts = +2 u32x4)

    const unsigned char* bb = &bstage[buf][0];
#pragma unroll
    for (int j = 0; j < 4; ++j) {
      ABfrag b;
      const u32x4* bp = (const u32x4*)(bb + j * 1024 + lane * 32);
      b.q[0] = bp[0];
      b.q[1] = bp[1];
      acc[j] = __builtin_amdgcn_wmma_f32_16x16x32_bf16(false, a.v, false, b.v,
                                                       (short)0, acc[j], false, false);
    }

    asm volatile("s_wait_asynccnt 0x0" ::: "memory");  // my fill of buf^1 done
    __syncthreads();                                   // everyone's fill done / reads done
  }
#undef ISSUE_STAGE

  const int mbase = blockIdx.x * 128 + wave * 16 + half * 8;
#pragma unroll
  for (int j = 0; j < 4; ++j) {
    int n = (nt0 + j) * 16 + l15;
    if (n >= NTOKN) continue;
    float bv = bias[n];
#pragma unroll
    for (int r = 0; r < 8; ++r)
      out[(size_t)(mbase + r) * NTOKN + n] = acc[j][r] + bv;
  }
}

// ---------------------------------------------------------------------------
extern "C" void kernel_launch(void* const* d_in, const int* in_sizes, int n_in,
                              void* d_out, int out_size, void* d_ws, size_t ws_size,
                              hipStream_t stream) {
  (void)in_sizes; (void)n_in; (void)out_size; (void)ws_size;
  const int*   input_seq = (const int*)  d_in[0];
  const float* hiddens   = (const float*)d_in[1];
  const float* embed     = (const float*)d_in[2];
  const float* out_bias  = (const float*)d_in[3];
  const float* W_zr1     = (const float*)d_in[4];
  const float* b_zr1     = (const float*)d_in[5];
  const float* W_g1      = (const float*)d_in[6];
  const float* b_g1      = (const float*)d_in[7];
  const float* W_zr2     = (const float*)d_in[8];
  const float* b_zr2     = (const float*)d_in[9];
  const float* W_g2      = (const float*)d_in[10];
  const float* b_g2      = (const float*)d_in[11];

  float* decoded = (float*)d_out;
  float* hT      = decoded + (size_t)S_LEN * BATCH * NTOKN;   // [L,B,D] tail

  char* ws = (char*)d_ws;
  size_t off = 0;
  auto alloc = [&](size_t bytes) -> void* {
    void* p = ws + off;
    off += (bytes + 255) & ~(size_t)255;
    return p;
  };

  unsigned int *Pzr1p[2], *Pg1p[2], *Pzr2p[2], *Pg2p[2];
  for (int l = 0; l < 2; ++l) {
    Pzr1p[l] = (unsigned int*)alloc((size_t)CATW * DIM * 2);
    Pg1p[l]  = (unsigned int*)alloc((size_t)CATW * H2D * 2);
    Pzr2p[l] = (unsigned int*)alloc((size_t)CATW * DIM * 2);
    Pg2p[l]  = (unsigned int*)alloc((size_t)CATW * H2D * 2);
  }
  unsigned int* PE = (unsigned int*)alloc((size_t)DIM * NPAD * 2);
  unsigned int* x0 = (unsigned int*)alloc((size_t)S_LEN * BATCH * DIM * 2);
  unsigned int* x1 = (unsigned int*)alloc((size_t)S_LEN * BATCH * DIM * 2);
  unsigned int* x2 = (unsigned int*)alloc((size_t)S_LEN * BATCH * DIM * 2);

  // --- pack weights (8 matrices) + embed^T to bf16 fragment order ---
  for (int l = 0; l < 2; ++l) {
    int dw_zr = (CATW * DIM) / 2, dw_g = (CATW * H2D) / 2;
    pack_b_kernel<<<(dw_zr + 255) / 256, 256, 0, stream>>>(
        W_zr1 + (size_t)l * DIM * CATW, Pzr1p[l], CATW, DIM, DIM);
    pack_b_kernel<<<(dw_g + 255) / 256, 256, 0, stream>>>(
        W_g1 + (size_t)l * H2D * CATW, Pg1p[l], CATW, H2D, H2D);
    pack_b_kernel<<<(dw_zr + 255) / 256, 256, 0, stream>>>(
        W_zr2 + (size_t)l * DIM * CATW, Pzr2p[l], CATW, DIM, DIM);
    pack_b_kernel<<<(dw_g + 255) / 256, 256, 0, stream>>>(
        W_g2 + (size_t)l * H2D * CATW, Pg2p[l], CATW, H2D, H2D);
  }
  {
    int dw_e = (DIM * NPAD) / 2;
    pack_b_kernel<<<(dw_e + 255) / 256, 256, 0, stream>>>(embed, PE, DIM, NPAD, NTOKN);
  }

  // --- embedding gather ---
  {
    int total_dw = S_LEN * BATCH * (DIM / 2);
    gather_kernel<<<(total_dw + 255) / 256, 256, 0, stream>>>(input_seq, embed, x0, total_dw);
  }

  // --- recurrent layers (sequential dependency) ---
  rnn_layer_kernel<<<1, 512, 0, stream>>>(
      x0, x1, hiddens,
      Pzr1p[0], b_zr1, Pg1p[0], b_g1, Pzr2p[0], b_zr2, Pg2p[0], b_g2,
      hT);
  rnn_layer_kernel<<<1, 512, 0, stream>>>(
      x1, x2, hiddens + (size_t)BATCH * DIM,
      Pzr1p[1], b_zr1 + DIM, Pg1p[1], b_g1 + H2D,
      Pzr2p[1], b_zr2 + DIM, Pg2p[1], b_g2 + H2D,
      hT + (size_t)BATCH * DIM);

  // --- decoder GEMM (dominant): 4096 x 33278, K=512, bf16 WMMA ---
  {
    dim3 grid((S_LEN * BATCH) / 128, NPAD / 64);
    decoder_kernel<<<grid, 256, 0, stream>>>((const unsigned short*)x2, PE, out_bias, decoded);
  }
}
